// AttentionCritic_61572651155854
// MI455X (gfx1250) — compile-verified, matrix operand order checked by reading, not run on hardware
//
#include <hip/hip_runtime.h>

#define N_AG   16
#define BATCH  2048
#define SDIM_  64
#define ADIM_  16
#define HDIM   128
#define HEADS_ 4
#define DHEAD  32
#define TPB    256

typedef __attribute__((ext_vector_type(16))) _Float16 v16h;
typedef __attribute__((ext_vector_type(8)))  _Float16 v8h;
typedef __attribute__((ext_vector_type(8)))  float    v8f;

__device__ __forceinline__ float leaky(float x) { return x > 0.0f ? x : 0.01f * x; }

// CDNA5 async global->LDS copy (ASYNCcnt-tracked), 16B per lane per issue.
// vdst VGPR holds the LDS byte offset (low 32 bits of the generic LDS address).
__device__ __forceinline__ void async_ld_lds_b128(void* lds, const void* gaddr) {
  asm volatile("global_load_async_to_lds_b128 %0, %1, off"
               :
               : "v"((unsigned int)(uintptr_t)lds),
                 "v"((unsigned long long)(uintptr_t)gaddr)
               : "memory");
}
__device__ __forceinline__ void wait_async0() {
  asm volatile("s_wait_asynccnt 0" ::: "memory");
}

// Build a 16-half WMMA fragment from LDS following the CDNA5 16-bit A/B layout:
// lanes 0-15  -> row, K = {kc+0..7, kc+16..23}
// lanes 16-31 -> row, K = {kc+8..15, kc+24..31}
// caller passes koff = kc + ((lane>>4)<<3); both 16B chunks are contiguous.
__device__ __forceinline__ v16h ld_frag(const _Float16* base, int ld, int row, int koff) {
  union { v16h v; v8h h[2]; } u;
  const _Float16* p = base + row * ld + koff;
  u.h[0] = *(const v8h*)(p);
  u.h[1] = *(const v8h*)(p + 16);
  return u.v;
}

// One wave computes a 64(M) x 16(N) strip of C: 4 accumulator tiles.
// sBt is B stored column-major ([n-col][k]) so B frags load like A frags.
__device__ __forceinline__ void gemm_strip4(const _Float16* sA, int lda,
                                            const _Float16* sBt, int ldb,
                                            int nc, int lrow, int kh,
                                            int KT, v8f acc[4]) {
  for (int kt = 0; kt < KT; ++kt) {
    const int kc = kt * 32 + kh;
    v16h bf = ld_frag(sBt, ldb, nc + lrow, kc);
#pragma unroll
    for (int mt = 0; mt < 4; ++mt) {
      v16h af = ld_frag(sA, lda, mt * 16 + lrow, kc);
      acc[mt] = __builtin_amdgcn_wmma_f32_16x16x32_f16(false, af, false, bf,
                                                       (short)0, acc[mt], false, false);
    }
  }
}

// ---------------------------------------------------------------------------
// Kernel 1: enc/senc + per-head keys/sels/vals projections, all WMMA.
// grid = (BATCH/64, N_AG), block = 256 (8 waves; wave w owns output cols 16w..)
// ---------------------------------------------------------------------------
__global__ __launch_bounds__(TPB) void k_encode(
    const float* __restrict__ states, const float* __restrict__ actions,
    const float* __restrict__ W_enc, const float* __restrict__ b_enc,
    const float* __restrict__ W_s,   const float* __restrict__ b_s,
    const float* __restrict__ Wk,    const float* __restrict__ Wq,
    const float* __restrict__ Wv,    const float* __restrict__ bv,
    _Float16* __restrict__ senc_ws, _Float16* __restrict__ keys_ws,
    _Float16* __restrict__ sels_ws, _Float16* __restrict__ vals_ws)
{
  __shared__ __align__(16) _Float16 sA[64 * 96];      // 12 KB activations (K padded to 96)
  __shared__ __align__(16) _Float16 sW[HDIM * HDIM];  // 32 KB weights, [col][k]
  __shared__ __align__(16) _Float16 sE[64 * HDIM];    // 16 KB enc (later senc) tile

  const int n    = blockIdx.y;
  const int b0   = blockIdx.x * 64;
  const int tid  = threadIdx.x;
  const int lane = tid & 31;
  const int wv   = tid >> 5;
  const int lrow = lane & 15;
  const int kh   = (lane >> 4) << 3;   // K-phase offset (0/8) == C row offset (0/8)
  const int nc   = wv * 16;

  // ---- stage 1: enc = leaky(sa @ W_enc + b_enc), kept in LDS -------------
  for (int i = tid; i < 64 * 96; i += TPB) {
    int r = i / 96, k = i - r * 96;
    float v = 0.0f;
    if (k < SDIM_)              v = states [((size_t)n * BATCH + b0 + r) * SDIM_ + k];
    else if (k < SDIM_ + ADIM_) v = actions[((size_t)n * BATCH + b0 + r) * ADIM_ + (k - SDIM_)];
    sA[r * 96 + k] = (_Float16)v;
  }
  for (int i = tid; i < HDIM * 96; i += TPB) {
    int h = i / 96, k = i - h * 96;
    float v = (k < 80) ? W_enc[((size_t)n * 80 + k) * HDIM + h] : 0.0f;
    sW[h * 96 + k] = (_Float16)v;
  }
  __syncthreads();
  {
    v8f acc[4] = {};
    gemm_strip4(sA, 96, sW, 96, nc, lrow, kh, 3, acc);
    const float bias = b_enc[n * HDIM + nc + lrow];
#pragma unroll
    for (int mt = 0; mt < 4; ++mt)
#pragma unroll
      for (int i = 0; i < 8; ++i)
        sE[(mt * 16 + i + kh) * HDIM + nc + lrow] = (_Float16)leaky(acc[mt][i] + bias);
  }
  __syncthreads();

  // ---- stage 2: keys = enc @ Wk_all  (cols packed e*32+d) ----------------
  for (int i = tid; i < HDIM * HDIM; i += TPB) {
    int c = i >> 7, k = i & 127;
    sW[c * HDIM + k] = (_Float16)Wk[((size_t)(c >> 5) * HDIM + k) * DHEAD + (c & 31)];
  }
  __syncthreads();
  {
    v8f acc[4] = {};
    gemm_strip4(sE, HDIM, sW, HDIM, nc, lrow, kh, 4, acc);
    const int e = (nc + lrow) >> 5, d = (nc + lrow) & 31;
#pragma unroll
    for (int mt = 0; mt < 4; ++mt)
#pragma unroll
      for (int i = 0; i < 8; ++i)
        keys_ws[(((size_t)e * N_AG + n) * BATCH + b0 + mt * 16 + i + kh) * DHEAD + d] =
            (_Float16)acc[mt][i];
  }
  __syncthreads();

  // ---- stage 3: vals = leaky(enc @ Wv_all + bv) --------------------------
  for (int i = tid; i < HDIM * HDIM; i += TPB) {
    int c = i >> 7, k = i & 127;
    sW[c * HDIM + k] = (_Float16)Wv[((size_t)(c >> 5) * HDIM + k) * DHEAD + (c & 31)];
  }
  __syncthreads();
  {
    v8f acc[4] = {};
    gemm_strip4(sE, HDIM, sW, HDIM, nc, lrow, kh, 4, acc);
    const float bias = bv[nc + lrow];   // (HEADS,D) flat == e*32+d
    const int e = (nc + lrow) >> 5, d = (nc + lrow) & 31;
#pragma unroll
    for (int mt = 0; mt < 4; ++mt)
#pragma unroll
      for (int i = 0; i < 8; ++i)
        vals_ws[(((size_t)e * N_AG + n) * BATCH + b0 + mt * 16 + i + kh) * DHEAD + d] =
            (_Float16)leaky(acc[mt][i] + bias);
  }
  __syncthreads();

  // ---- stage 4: senc = leaky(states @ W_s + b_s) -> sE + global ----------
  for (int i = tid; i < 64 * 96; i += TPB) {
    int r = i / 96, k = i - r * 96;
    float v = (k < SDIM_) ? states[((size_t)n * BATCH + b0 + r) * SDIM_ + k] : 0.0f;
    sA[r * 96 + k] = (_Float16)v;
  }
  for (int i = tid; i < HDIM * 96; i += TPB) {
    int h = i / 96, k = i - h * 96;
    float v = (k < SDIM_) ? W_s[((size_t)n * SDIM_ + k) * HDIM + h] : 0.0f;
    sW[h * 96 + k] = (_Float16)v;
  }
  __syncthreads();
  {
    v8f acc[4] = {};
    gemm_strip4(sA, 96, sW, 96, nc, lrow, kh, 2, acc);
    const float bias = b_s[n * HDIM + nc + lrow];
#pragma unroll
    for (int mt = 0; mt < 4; ++mt)
#pragma unroll
      for (int i = 0; i < 8; ++i) {
        float v = leaky(acc[mt][i] + bias);
        int row = mt * 16 + i + kh;
        sE[row * HDIM + nc + lrow] = (_Float16)v;
        senc_ws[((size_t)n * BATCH + b0 + row) * HDIM + nc + lrow] = (_Float16)v;
      }
  }
  __syncthreads();

  // ---- stage 5: sels = senc @ Wq_all --------------------------------------
  for (int i = tid; i < HDIM * HDIM; i += TPB) {
    int c = i >> 7, k = i & 127;
    sW[c * HDIM + k] = (_Float16)Wq[((size_t)(c >> 5) * HDIM + k) * DHEAD + (c & 31)];
  }
  __syncthreads();
  {
    v8f acc[4] = {};
    gemm_strip4(sE, HDIM, sW, HDIM, nc, lrow, kh, 4, acc);
    const int e = (nc + lrow) >> 5, d = (nc + lrow) & 31;
#pragma unroll
    for (int mt = 0; mt < 4; ++mt)
#pragma unroll
      for (int i = 0; i < 8; ++i)
        sels_ws[(((size_t)e * N_AG + n) * BATCH + b0 + mt * 16 + i + kh) * DHEAD + d] =
            (_Float16)acc[mt][i];
  }
}

// ---------------------------------------------------------------------------
// Kernel 2: mixed-logit softmax attention. One thread per (head, agent, b).
// ---------------------------------------------------------------------------
__global__ __launch_bounds__(TPB) void k_attn(
    const float* __restrict__ clst,
    const _Float16* __restrict__ keys_ws, const _Float16* __restrict__ sels_ws,
    const _Float16* __restrict__ vals_ws, _Float16* __restrict__ attn_ws)
{
  const int tid = blockIdx.x * TPB + threadIdx.x;
  const int e   = tid / (N_AG * BATCH);
  const int rem = tid - e * (N_AG * BATCH);
  const int n   = rem / BATCH;
  const int b   = rem - n * BATCH;

  float sel[DHEAD];
  {
    const _Float16* sp = sels_ws + (((size_t)e * N_AG + n) * BATCH + b) * DHEAD;
#pragma unroll
    for (int c = 0; c < 4; ++c) {
      v8h t = *(const v8h*)(sp + c * 8);
#pragma unroll
      for (int i = 0; i < 8; ++i) sel[c * 8 + i] = (float)t[i];
    }
  }

  const float inv_sqrt_d = 0.17677669529663687f;  // 1/sqrt(32)
  float logits[N_AG - 1];
#pragma unroll 1
  for (int j = 0; j < N_AG - 1; ++j) {
    const int m = (j < n) ? j : j + 1;
    const _Float16* kp = keys_ws + (((size_t)e * N_AG + m) * BATCH + b) * DHEAD;
    __builtin_prefetch(vals_ws + (((size_t)e * N_AG + m) * BATCH + b) * DHEAD, 0, 1);
    float acc = 0.0f;
#pragma unroll
    for (int c = 0; c < 4; ++c) {
      v8h t = *(const v8h*)(kp + c * 8);
#pragma unroll
      for (int i = 0; i < 8; ++i) acc += sel[c * 8 + i] * (float)t[i];
    }
    const float cl = clst[(((size_t)n * HEADS_ + e) * BATCH + b) * (N_AG - 1) + j];
    logits[j] = 0.5f * cl + 0.5f * (acc * inv_sqrt_d);
  }

  float mx = logits[0];
#pragma unroll
  for (int j = 1; j < N_AG - 1; ++j) mx = fmaxf(mx, logits[j]);
  float s = 0.0f;
#pragma unroll
  for (int j = 0; j < N_AG - 1; ++j) { logits[j] = __expf(logits[j] - mx); s += logits[j]; }
  const float rs = 1.0f / s;

  float outv[DHEAD];
#pragma unroll
  for (int d = 0; d < DHEAD; ++d) outv[d] = 0.0f;
#pragma unroll 1
  for (int j = 0; j < N_AG - 1; ++j) {
    const int m = (j < n) ? j : j + 1;
    const _Float16* vp = vals_ws + (((size_t)e * N_AG + m) * BATCH + b) * DHEAD;
    const float w = logits[j] * rs;
#pragma unroll
    for (int c = 0; c < 4; ++c) {
      v8h t = *(const v8h*)(vp + c * 8);
#pragma unroll
      for (int i = 0; i < 8; ++i) outv[c * 8 + i] += w * (float)t[i];
    }
  }
  _Float16* ap = attn_ws + ((size_t)n * BATCH + b) * HDIM + e * DHEAD;
#pragma unroll
  for (int d = 0; d < DHEAD; ++d) ap[d] = (_Float16)outv[d];
}

// ---------------------------------------------------------------------------
// Kernel 3: h1 = leaky([senc|attn] @ Wc1 + bc1), K=256 processed in 2 chunks.
// cin tile is pure f16 and contiguous in global -> async global->LDS copy
// overlapped with the f32->f16 weight conversion loads.
// grid = (BATCH/64, N_AG), block = 256
// ---------------------------------------------------------------------------
__global__ __launch_bounds__(TPB) void k_critic1(
    const float* __restrict__ Wc1, const float* __restrict__ bc1,
    const _Float16* __restrict__ senc_ws, const _Float16* __restrict__ attn_ws,
    _Float16* __restrict__ h1_ws)
{
  __shared__ __align__(16) _Float16 sCin[64 * HDIM];   // 16 KB (one K-half of cin)
  __shared__ __align__(16) _Float16 sW1[HDIM * HDIM];  // 32 KB (one K-half of Wc1, [h][k])

  const int n    = blockIdx.y;
  const int b0   = blockIdx.x * 64;
  const int tid  = threadIdx.x;
  const int lane = tid & 31;
  const int wv   = tid >> 5;
  const int lrow = lane & 15;
  const int kh   = (lane >> 4) << 3;
  const int nc   = wv * 16;

  v8f acc[4] = {};
  for (int p = 0; p < 2; ++p) {
    // async stream: 64x128 f16 tile, contiguous in global (1024 x 16B chunks)
    const _Float16* src = (p ? attn_ws : senc_ws) + ((size_t)n * BATCH + b0) * HDIM;
    for (int c = tid; c < (64 * HDIM) / 8; c += TPB)
      async_ld_lds_b128(sCin + c * 8, src + c * 8);
    // VALU stream: transpose + f32->f16 one K-half of Wc1
    for (int i = tid; i < HDIM * HDIM; i += TPB) {
      int h = i >> 7, k = i & 127;
      sW1[h * HDIM + k] = (_Float16)Wc1[((size_t)n * 256 + p * HDIM + k) * HDIM + h];
    }
    wait_async0();
    __syncthreads();
    gemm_strip4(sCin, HDIM, sW1, HDIM, nc, lrow, kh, 4, acc);
    __syncthreads();
  }
  const float bias = bc1[n * HDIM + nc + lrow];
#pragma unroll
  for (int mt = 0; mt < 4; ++mt)
#pragma unroll
    for (int i = 0; i < 8; ++i)
      h1_ws[((size_t)n * BATCH + b0 + mt * 16 + i + kh) * HDIM + nc + lrow] =
          (_Float16)leaky(acc[mt][i] + bias);
}

// ---------------------------------------------------------------------------
// Kernel 4: all_q = h1 @ Wc2 + bc2 (N=16), then q = all_q[argmax(actions)].
// h1 tile is pure f16 and contiguous -> async global->LDS copy.
// grid = (BATCH/128, N_AG), block = 256 (wave w owns output rows 16w..16w+15)
// ---------------------------------------------------------------------------
__global__ __launch_bounds__(TPB) void k_critic2(
    const float* __restrict__ actions,
    const float* __restrict__ Wc2, const float* __restrict__ bc2,
    const _Float16* __restrict__ h1_ws, float* __restrict__ out)
{
  __shared__ __align__(16) _Float16 sH1[128 * HDIM];  // 32 KB
  __shared__ __align__(16) _Float16 sW2[16 * HDIM];   // 4 KB, [a][k]
  __shared__ float sQ[128 * 16];                      // 8 KB

  const int n    = blockIdx.y;
  const int b0   = blockIdx.x * 128;
  const int tid  = threadIdx.x;
  const int lane = tid & 31;
  const int wv   = tid >> 5;
  const int lrow = lane & 15;
  const int kh   = (lane >> 4) << 3;

  // async stream: 128x128 f16 tile, contiguous (2048 x 16B chunks)
  {
    const _Float16* src = h1_ws + ((size_t)n * BATCH + b0) * HDIM;
    for (int c = tid; c < (128 * HDIM) / 8; c += TPB)
      async_ld_lds_b128(sH1 + c * 8, src + c * 8);
  }
  // VALU stream: transpose + f32->f16 Wc2
  for (int i = tid; i < 16 * HDIM; i += TPB) {
    int a = i >> 7, k = i & 127;
    sW2[a * HDIM + k] = (_Float16)Wc2[((size_t)n * HDIM + k) * 16 + a];
  }
  wait_async0();
  __syncthreads();

  {
    v8f acc = {};
    for (int kt = 0; kt < 4; ++kt) {
      const int kc = kt * 32 + kh;
      v16h bf = ld_frag(sW2, HDIM, lrow, kc);
      v16h af = ld_frag(sH1, HDIM, wv * 16 + lrow, kc);
      acc = __builtin_amdgcn_wmma_f32_16x16x32_f16(false, af, false, bf,
                                                   (short)0, acc, false, false);
    }
    const float bias = bc2[n * 16 + lrow];
#pragma unroll
    for (int i = 0; i < 8; ++i)
      sQ[(wv * 16 + i + kh) * 16 + lrow] = acc[i] + bias;
  }
  __syncthreads();

  if (tid < 128) {
    const int b = b0 + tid;
    const float* ap = actions + ((size_t)n * BATCH + b) * ADIM_;
    int amax = 0;
    float mv = ap[0];
#pragma unroll
    for (int a = 1; a < ADIM_; ++a) {
      float v = ap[a];
      if (v > mv) { mv = v; amax = a; }
    }
    out[(size_t)n * BATCH + b] = sQ[tid * 16 + amax];
  }
}

// ---------------------------------------------------------------------------
extern "C" void kernel_launch(void* const* d_in, const int* in_sizes, int n_in,
                              void* d_out, int out_size, void* d_ws, size_t ws_size,
                              hipStream_t stream) {
  (void)in_sizes; (void)n_in; (void)out_size; (void)ws_size;
  const float* states  = (const float*)d_in[0];
  const float* actions = (const float*)d_in[1];
  const float* clst    = (const float*)d_in[2];
  const float* W_enc   = (const float*)d_in[3];
  const float* b_enc   = (const float*)d_in[4];
  const float* W_s     = (const float*)d_in[5];
  const float* b_s     = (const float*)d_in[6];
  const float* Wk      = (const float*)d_in[7];
  const float* Wq      = (const float*)d_in[8];
  const float* Wv      = (const float*)d_in[9];
  const float* bv      = (const float*)d_in[10];
  const float* Wc1     = (const float*)d_in[11];
  const float* bc1     = (const float*)d_in[12];
  const float* Wc2     = (const float*)d_in[13];
  const float* bc2     = (const float*)d_in[14];
  float* out = (float*)d_out;

  const size_t SZ = (size_t)N_AG * BATCH * HDIM;  // 4,194,304 halves per buffer
  _Float16* ws      = (_Float16*)d_ws;
  _Float16* senc_ws = ws;
  _Float16* keys_ws = ws + 1 * SZ;
  _Float16* sels_ws = ws + 2 * SZ;
  _Float16* vals_ws = ws + 3 * SZ;
  _Float16* attn_ws = ws + 4 * SZ;
  _Float16* h1_ws   = keys_ws;  // keys are dead after k_attn; alias to cap ws at 40 MB

  k_encode<<<dim3(BATCH / 64, N_AG), TPB, 0, stream>>>(
      states, actions, W_enc, b_enc, W_s, b_s, Wk, Wq, Wv, bv,
      senc_ws, keys_ws, sels_ws, vals_ws);

  k_attn<<<(HEADS_ * N_AG * BATCH) / TPB, TPB, 0, stream>>>(
      clst, keys_ws, sels_ws, vals_ws, attn_ws);

  k_critic1<<<dim3(BATCH / 64, N_AG), TPB, 0, stream>>>(
      Wc1, bc1, senc_ws, attn_ws, h1_ws);

  k_critic2<<<dim3(BATCH / 128, N_AG), TPB, 0, stream>>>(
      actions, Wc2, bc2, h1_ws, out);
}
